// MPM_1022202216465
// MI455X (gfx1250) — matrix operand order repaired and checked by reading, not compiled, not run on hardware
//
#include <hip/hip_runtime.h>

// ---------------------------------------------------------------------------
// MI455X / gfx1250 implementation of the multi-scale mask-prediction network.
// Conv encoders + proto pooling run as implicit-GEMM on v_wmma_f32_16x16x32_f16
// (f16 operands, f32 accumulate; GroupNorm renormalizes each layer so f16 is
// numerically adequate). Proto stages operand tiles into LDS with the Tensor
// Data Mover (tensor_load_to_lds + s_wait_tensorcnt) when the builtin exists.
// ---------------------------------------------------------------------------

typedef __attribute__((ext_vector_type(16))) _Float16 v16h;
typedef __attribute__((ext_vector_type(8)))  _Float16 v8h;
typedef __attribute__((ext_vector_type(8)))  float    v8f;
typedef __attribute__((ext_vector_type(4)))  unsigned int uint4v;
typedef __attribute__((ext_vector_type(8)))  int      int8v;
typedef __attribute__((ext_vector_type(4)))  int      int4v;

#define BATCH 8
#define NINST 64
#define SRC_H 256
#define SRC_W 256
#define EPS_GN 1e-5f
#define EPS_AREA 1e-7f

#ifdef __has_builtin
#if __has_builtin(__builtin_amdgcn_tensor_load_to_lds)
#define HAVE_TDM 1
#endif
#endif
#ifndef HAVE_TDM
#define HAVE_TDM 0
#endif

#if HAVE_TDM
// Issue one 2D TDM load: tile (tile_d1 rows x tile_d0 cols) of a row-major
// f16 tensor with row stride stride0 (elements), into LDS at lds_addr.
__device__ __forceinline__ void tdm_load_2d(unsigned lds_addr, const void* gaddr,
                                            int tensor_d0, int tensor_d1,
                                            int tile_d0, int tile_d1, int stride0) {
  unsigned long long ga = (unsigned long long)(uintptr_t)gaddr;
  uint4v g0;
  g0.x = 1u;                                            // count=1, flags=0
  g0.y = lds_addr;                                      // LDS byte address
  g0.z = (unsigned)(ga & 0xFFFFFFFFu);                  // global_addr[31:0]
  g0.w = (unsigned)((ga >> 32) & 0x01FFFFFFu) | (2u << 30); // addr[56:32] | type=2
  int8v g1;
  g1[0] = 0x10000;                                      // data_size=1 (2 bytes)
  g1[1] = (int)(((unsigned)tensor_d0 & 0xFFFFu) << 16); // tensor_dim0 lo16
  g1[2] = (int)((((unsigned)tensor_d0 >> 16) & 0xFFFFu) |
                (((unsigned)tensor_d1 & 0xFFFFu) << 16));
  g1[3] = (int)((((unsigned)tensor_d1 >> 16) & 0xFFFFu) |
                (((unsigned)tile_d0 & 0xFFFFu) << 16)); // tile_dim0
  g1[4] = (int)((unsigned)tile_d1 & 0xFFFFu);           // tile_dim1
  g1[5] = stride0;                                      // tensor_dim0_stride lo32
  g1[6] = 0;
  g1[7] = 0;
  int4v gz = {0, 0, 0, 0};
#if __clang_major__ >= 23
  int8v gz8 = {0, 0, 0, 0, 0, 0, 0, 0};
  __builtin_amdgcn_tensor_load_to_lds(g0, g1, gz, gz, gz8, 0);
#else
  __builtin_amdgcn_tensor_load_to_lds(g0, g1, gz, gz, 0);
#endif
}
#endif

// ---------------------------------------------------------------------------
// Pack conv weights (Cout, Cin, 3, 3) f32 -> f16 in a contiguous-k WMMA A-fragment
// lane layout for 16x16x32: lane L holds row m = L&15; half j maps to
// k_local = (L>>4)*16 + j (B is gathered in the same k order).
// ---------------------------------------------------------------------------
__global__ void k_pack_w(const float* __restrict__ w, _Float16* __restrict__ dst,
                         int Cout, int Cin) {
  int K = Cin * 9;
  int Ksteps = (K + 31) >> 5;
  int total = (Cout >> 4) * Ksteps * 512;
  for (int idx = blockIdx.x * blockDim.x + threadIdx.x; idx < total;
       idx += gridDim.x * blockDim.x) {
    int j  = idx & 15;
    int L  = (idx >> 4) & 31;
    int t  = idx >> 9;
    int ks = t % Ksteps;
    int mtile = t / Ksteps;
    int khi = L >> 4;
    int kl  = khi * 16 + j;
    int Kg  = ks * 32 + kl;
    int m   = mtile * 16 + (L & 15);
    float v = (Kg < K) ? w[m * K + Kg] : 0.0f;
    dst[idx] = (_Float16)v;
  }
}

__global__ void k_cvt_f16(const float* __restrict__ in, _Float16* __restrict__ out, int n) {
  for (int i = blockIdx.x * blockDim.x + threadIdx.x; i < n; i += gridDim.x * blockDim.x)
    out[i] = (_Float16)in[i];
}

// ---------------------------------------------------------------------------
// Implicit-GEMM 3x3 SAME conv. One wave computes a 16(cout) x 32(pixel) tile:
// one A fragment feeds two WMMAs per K-step (2x arithmetic intensity on the
// weight stream).
// ---------------------------------------------------------------------------
__global__ void k_conv_wmma(const _Float16* __restrict__ act,   // (B,Cin,H,W)
                            const _Float16* __restrict__ wpack, // packed A
                            const float* __restrict__ bias,
                            _Float16* __restrict__ out,         // (B,Cout,H,W)
                            int Bn, int Cin, int Cout, int H, int W) {
  const int N = H * W;
  const int Kreal = Cin * 9;
  const int Ksteps = (Kreal + 31) >> 5;
  const int Mtiles = Cout >> 4;
  const int Ntiles2 = N >> 5;                // 32 pixels per wave
  const int tilesPerImg = Mtiles * Ntiles2;
  const int total = Bn * tilesPerImg;

  int wid = blockIdx.x * (blockDim.x >> 5) + (threadIdx.x >> 5);
  if (wid >= total) return;                  // wave-uniform
  int lane = threadIdx.x & 31;
  int b    = wid / tilesPerImg;
  int rem  = wid - b * tilesPerImg;
  int mtile = rem / Ntiles2;
  int nt2   = rem - mtile * Ntiles2;

  int col = lane & 15;
  int khi = lane >> 4;
  int p0 = nt2 * 32 + col;
  int p1 = p0 + 16;
  int y0 = p0 / W, x0 = p0 - y0 * W;
  int y1 = p1 / W, x1 = p1 - y1 * W;

  const _Float16* actb  = act + (size_t)b * Cin * N;
  const _Float16* wbase = wpack + (size_t)mtile * Ksteps * 512;

  v8f acc0 = {}, acc1 = {};
  for (int ks = 0; ks < Ksteps; ++ks) {
    v16h afrag = *(const v16h*)(wbase + ((size_t)ks * 32 + lane) * 16);
    if (ks + 1 < Ksteps)
      __builtin_prefetch(wbase + ((size_t)(ks + 1) * 32 + lane) * 16, 0, 1);

    v16h bf0, bf1;
    int Kg  = ks * 32 + khi * 16;
    int cin = Kg / 9;
    int r   = Kg - cin * 9;
#pragma unroll
    for (int j = 0; j < 16; ++j) {
      _Float16 v0 = (_Float16)0.0f, v1 = (_Float16)0.0f;
      if (Kg < Kreal) {
        int ky = r / 3, kx = r - ky * 3;
        int ya = y0 + ky - 1, xa = x0 + kx - 1;
        int yb = y1 + ky - 1, xb = x1 + kx - 1;
        const _Float16* cb = actb + (size_t)cin * N;
        if (ya >= 0 && ya < H && xa >= 0 && xa < W) v0 = cb[ya * W + xa];
        if (yb >= 0 && yb < H && xb >= 0 && xb < W) v1 = cb[yb * W + xb];
      }
      bf0[j] = v0;
      bf1[j] = v1;
      ++Kg; ++r;
      if (r == 9) { r = 0; ++cin; }
    }
    acc0 = __builtin_amdgcn_wmma_f32_16x16x32_f16(false, afrag, false, bf0,
                                                  (short)0, acc0, false, false);
    acc1 = __builtin_amdgcn_wmma_f32_16x16x32_f16(false, afrag, false, bf1,
                                                  (short)0, acc1, false, false);
  }

  _Float16* ob = out + (size_t)b * Cout * N;
#pragma unroll
  for (int rr = 0; rr < 8; ++rr) {
    int co = mtile * 16 + rr + 8 * khi;
    float bs = bias[co];
    ob[(size_t)co * N + p0] = (_Float16)(acc0[rr] + bs);
    ob[(size_t)co * N + p1] = (_Float16)(acc1[rr] + bs);
  }
}

// ---------------------------------------------------------------------------
// GroupNorm statistics: one block per (image, group); wide f16 loads.
// ---------------------------------------------------------------------------
__global__ void k_gn_stats(const _Float16* __restrict__ x, float* __restrict__ murs,
                           int C, int N) {
  int bg = blockIdx.x;
  int b = bg >> 2, g = bg & 3;
  int cpg = C >> 2;
  int cnt = cpg * N;
  const _Float16* p = x + ((size_t)b * C + (size_t)g * cpg) * N;
  float s = 0.f, s2 = 0.f;
  int cnt8 = cnt >> 3;
  for (int i = threadIdx.x; i < cnt8; i += blockDim.x) {
    v8h v = *(const v8h*)(p + (size_t)i * 8);
#pragma unroll
    for (int j = 0; j < 8; ++j) { float f = (float)v[j]; s += f; s2 += f * f; }
  }
  __shared__ float sa[256], sb[256];
  sa[threadIdx.x] = s; sb[threadIdx.x] = s2;
  __syncthreads();
  for (int off = 128; off > 0; off >>= 1) {
    if (threadIdx.x < off) {
      sa[threadIdx.x] += sa[threadIdx.x + off];
      sb[threadIdx.x] += sb[threadIdx.x + off];
    }
    __syncthreads();
  }
  if (threadIdx.x == 0) {
    float mu  = sa[0] / (float)cnt;
    float var = sb[0] / (float)cnt - mu * mu;
    murs[bg * 2 + 0] = mu;
    murs[bg * 2 + 1] = rsqrtf(var + EPS_GN);
  }
}

__global__ void k_gn_apply(const _Float16* __restrict__ x, const float* __restrict__ murs,
                           const float* __restrict__ gam, const float* __restrict__ bet,
                           _Float16* __restrict__ yout, int Bn, int C, int N) {
  int total = Bn * C * N;
  int cpg = C >> 2;
  for (int idx = blockIdx.x * blockDim.x + threadIdx.x; idx < total;
       idx += gridDim.x * blockDim.x) {
    int c = (idx / N) % C;
    int b = idx / (C * N);
    int g = c / cpg;
    float mu = murs[(b * 4 + g) * 2 + 0];
    float rs = murs[(b * 4 + g) * 2 + 1];
    float v = ((float)x[idx] - mu) * rs;
    v = v * gam[c] + bet[c];
    yout[idx] = (_Float16)fmaxf(v, 0.0f);
  }
}

// ---------------------------------------------------------------------------
// Bilinear align_corners resize of instances -> (B,64,Hs,Ws) f16.
// ---------------------------------------------------------------------------
__global__ void k_resize(const float* __restrict__ ins, _Float16* __restrict__ outr,
                         int Hs, int Ws) {
  int total = BATCH * NINST * Hs * Ws;
  float sy = (float)(SRC_H - 1) / (float)(Hs - 1);
  float sx = (float)(SRC_W - 1) / (float)(Ws - 1);
  for (int idx = blockIdx.x * blockDim.x + threadIdx.x; idx < total;
       idx += gridDim.x * blockDim.x) {
    int xw = idx % Ws;
    int t  = idx / Ws;
    int yh = t % Hs;
    int q  = t / Hs;
    float fy = yh * sy;  int iy0 = (int)fy;  int iy1 = iy0 + 1 < SRC_H ? iy0 + 1 : SRC_H - 1;
    float wy = fy - iy0;
    float fx = xw * sx;  int ix0 = (int)fx;  int ix1 = ix0 + 1 < SRC_W ? ix0 + 1 : SRC_W - 1;
    float wx = fx - ix0;
    const float* base = ins + (size_t)q * SRC_H * SRC_W;
    float v00 = base[iy0 * SRC_W + ix0], v01 = base[iy0 * SRC_W + ix1];
    float v10 = base[iy1 * SRC_W + ix0], v11 = base[iy1 * SRC_W + ix1];
    float v = v00 * (1.f - wy) * (1.f - wx) + v01 * (1.f - wy) * wx
            + v10 * wy * (1.f - wx)         + v11 * wy * wx;
    outr[idx] = (_Float16)v;
  }
}

// One block per (b, inst): area = sum(resized mask) + eps.
__global__ void k_area(const _Float16* __restrict__ insr, float* __restrict__ area, int N) {
  const _Float16* p = insr + (size_t)blockIdx.x * N;
  float s = 0.f;
  int n8 = N >> 3;
  for (int i = threadIdx.x; i < n8; i += blockDim.x) {
    v8h v = *(const v8h*)(p + (size_t)i * 8);
#pragma unroll
    for (int j = 0; j < 8; ++j) s += (float)v[j];
  }
  __shared__ float sd[256];
  sd[threadIdx.x] = s;
  __syncthreads();
  for (int off = 128; off > 0; off >>= 1) {
    if (threadIdx.x < off) sd[threadIdx.x] += sd[threadIdx.x + off];
    __syncthreads();
  }
  if (threadIdx.x == 0) area[blockIdx.x] = sd[0] + EPS_AREA;
}

// ---------------------------------------------------------------------------
// Proto: P[b][n][c] = sum_hw insr[b,n,hw] * feat[b,c,hw] / area[b,n].
// One block (8 waves) per image. K is processed in 128-pixel chunks staged in
// LDS: feat chunk 128x128 f16 (32KB) + ins chunk 64x128 f16 (16KB), loaded by
// the Tensor Data Mover when available (else cooperative wide copies).
// Wave w owns channel tile ntile=w and accumulates all 4 instance tiles.
// ---------------------------------------------------------------------------
__global__ void k_proto_wmma(const _Float16* __restrict__ insr, // (B,64,N)
                             const _Float16* __restrict__ feat, // (B,128,N)
                             const float* __restrict__ area,    // (B,64)
                             float* __restrict__ P,             // (B,64,128)
                             int N) {
  __shared__ _Float16 sFeat[128 * 128];
  __shared__ _Float16 sIns[64 * 128];
  int b    = blockIdx.x;
  int wid  = threadIdx.x >> 5;               // == ntile (0..7)
  int lane = threadIdx.x & 31;
  int col  = lane & 15, khi = lane >> 4;

  v8f acc[4] = {};
  int nchunks = N >> 7;
  for (int ch = 0; ch < nchunks; ++ch) {
#if HAVE_TDM
    if (threadIdx.x < 32) {
      tdm_load_2d((unsigned)(uintptr_t)sFeat,
                  feat + (size_t)b * 128 * N + ch * 128, N, 128, 128, 128, N);
      tdm_load_2d((unsigned)(uintptr_t)sIns,
                  insr + (size_t)b * NINST * N + ch * 128, N, 64, 128, 64, N);
      __builtin_amdgcn_s_wait_tensorcnt(0);
    }
    __syncthreads();
#else
    for (int t = threadIdx.x; t < 2048; t += 256) {  // 128 rows x 16 v8h
      int r = t >> 4, c = (t & 15) << 3;
      *(v8h*)(sFeat + r * 128 + c) =
          *(const v8h*)(feat + ((size_t)b * 128 + r) * N + ch * 128 + c);
    }
    for (int t = threadIdx.x; t < 1024; t += 256) {  // 64 rows x 16 v8h
      int r = t >> 4, c = (t & 15) << 3;
      *(v8h*)(sIns + r * 128 + c) =
          *(const v8h*)(insr + ((size_t)b * NINST + r) * N + ch * 128 + c);
    }
    __syncthreads();
#endif
    const _Float16* bbase = sFeat + (wid * 16 + col) * 128;
#pragma unroll
    for (int ks = 0; ks < 4; ++ks) {
      v16h bf = *(const v16h*)(bbase + ks * 32 + khi * 16);
#pragma unroll
      for (int mt = 0; mt < 4; ++mt) {
        const _Float16* arow = sIns + (mt * 16 + col) * 128 + ks * 32;
        v8h a0 = *(const v8h*)(arow + khi * 16);
        v8h a1 = *(const v8h*)(arow + khi * 16 + 8);
        v16h af;
#pragma unroll
        for (int j = 0; j < 8; ++j) { af[j] = a0[j]; af[j + 8] = a1[j]; }
        acc[mt] = __builtin_amdgcn_wmma_f32_16x16x32_f16(false, af, false, bf,
                                                         (short)0, acc[mt], false, false);
      }
    }
    __syncthreads();
  }

  float* pb = P + (size_t)b * NINST * 128;
#pragma unroll
  for (int mt = 0; mt < 4; ++mt) {
#pragma unroll
    for (int rr = 0; rr < 8; ++rr) {
      int inst = mt * 16 + rr + 8 * khi;
      pb[inst * 128 + wid * 16 + col] = acc[mt][rr] / area[b * NINST + inst];
    }
  }
}

// ---------------------------------------------------------------------------
// Generic small linear: out[b,i,j] = act( sum_k (in[b,i,k] (+in2)) * w[k,j] + bias[j] ).
// ---------------------------------------------------------------------------
__global__ void k_linear(const float* __restrict__ in, const float* __restrict__ in2,
                         const float* __restrict__ w, const float* __restrict__ bias,
                         float* __restrict__ out, int Bn, int M, int K, int Nout,
                         int do_relu) {
  int total = Bn * M * Nout;
  for (int idx = blockIdx.x * blockDim.x + threadIdx.x; idx < total;
       idx += gridDim.x * blockDim.x) {
    int j  = idx % Nout;
    int bi = idx / Nout;
    const float* ip = in + (size_t)bi * K;
    float s = bias[j];
    if (in2) {
      const float* ip2 = in2 + (size_t)bi * K;
      for (int k = 0; k < K; ++k) s += (ip[k] + ip2[k]) * w[k * Nout + j];
    } else {
      for (int k = 0; k < K; ++k) s += ip[k] * w[k * Nout + j];
    }
    if (do_relu) s = fmaxf(s, 0.0f);
    out[idx] = s;
  }
}

// Final 256 -> 1 layer: logits to ws, sigmoid(cats) straight to d_out.
__global__ void k_logit(const float* __restrict__ Z, const float* __restrict__ w,
                        const float* __restrict__ bias, float* __restrict__ logit,
                        float* __restrict__ cats, int K) {
  int total = BATCH * NINST;
  for (int idx = blockIdx.x * blockDim.x + threadIdx.x; idx < total;
       idx += gridDim.x * blockDim.x) {
    const float* zp = Z + (size_t)idx * K;
    float s = bias[0];
    for (int k = 0; k < K; ++k) s += zp[k] * w[k];
    logit[idx] = s;
    cats[idx] = 1.0f / (1.0f + __expf(-s));
  }
}

// masks[b,0,px] = 1 - m; masks[b,1,px] = m;  m = sum_q logit[b,q]*ins[b,q,px].
__global__ void k_mask(const float* __restrict__ logit, const float* __restrict__ ins,
                       float* __restrict__ outm) {
  const int NPIX = SRC_H * SRC_W;
  int total = BATCH * NPIX;
  for (int idx = blockIdx.x * blockDim.x + threadIdx.x; idx < total;
       idx += gridDim.x * blockDim.x) {
    int b  = idx / NPIX;
    int px = idx - b * NPIX;
    const float* lp = logit + b * NINST;
    const float* ip = ins + (size_t)b * NINST * NPIX + px;
    float m = 0.f;
#pragma unroll 8
    for (int q = 0; q < NINST; ++q) m += lp[q] * ip[(size_t)q * NPIX];
    outm[((size_t)b * 2 + 0) * NPIX + px] = 1.0f - m;
    outm[((size_t)b * 2 + 1) * NPIX + px] = m;
  }
}

// ---------------------------------------------------------------------------
// Host-side orchestration.
// ---------------------------------------------------------------------------
static inline dim3 gridFor(long long n, int blk) {
  long long g = (n + blk - 1) / blk;
  if (g < 1) g = 1;
  return dim3((unsigned)g);
}

extern "C" void kernel_launch(void* const* d_in, const int* in_sizes, int n_in,
                              void* d_out, int out_size, void* d_ws, size_t ws_size,
                              hipStream_t stream) {
  (void)in_sizes; (void)n_in; (void)out_size; (void)ws_size;
  const float* corr[3]    = {(const float*)d_in[0],   // corr4 (8,3,32,32)
                             (const float*)d_in[1],   // corr3 (8,6,64,64)
                             (const float*)d_in[2]};  // corr2 (8,4,128,128)
  const float* instances  = (const float*)d_in[3];    // (8,64,256,256)

  struct ConvP { const float *w, *b, *g, *be; int Cout, Cin; };
  ConvP enc[3][3];
  const int cin0[3] = {3, 6, 4};
  const int couts[3] = {16, 64, 128};
  int base = 4;
  for (int s = 0; s < 3; ++s) {
    int prev = cin0[s];
    for (int l = 0; l < 3; ++l) {
      enc[s][l].w  = (const float*)d_in[base++];
      enc[s][l].b  = (const float*)d_in[base++];
      enc[s][l].g  = (const float*)d_in[base++];
      enc[s][l].be = (const float*)d_in[base++];
      enc[s][l].Cout = couts[l];
      enc[s][l].Cin  = prev;
      prev = couts[l];
    }
  }
  const float* w1_43 = (const float*)d_in[base++]; const float* b1_43 = (const float*)d_in[base++];
  const float* w2_43 = (const float*)d_in[base++]; const float* b2_43 = (const float*)d_in[base++];
  const float* w1_32 = (const float*)d_in[base++]; const float* b1_32 = (const float*)d_in[base++];
  const float* w2_32 = (const float*)d_in[base++]; const float* b2_32 = (const float*)d_in[base++];
  const float* wd0 = (const float*)d_in[base++]; const float* bd0 = (const float*)d_in[base++];
  const float* wd1 = (const float*)d_in[base++]; const float* bd1 = (const float*)d_in[base++];
  const float* wd2 = (const float*)d_in[base++]; const float* bd2 = (const float*)d_in[base++];

  // ---- workspace layout -------------------------------------------------
  char* ws = (char*)d_ws;
  size_t off = 0;
  auto take = [&](size_t bytes) -> char* {
    char* p = ws + off;
    off += (bytes + 255) & ~(size_t)255;
    return p;
  };
  size_t woff[3][3];
  size_t wcur = 0;
  for (int s = 0; s < 3; ++s)
    for (int l = 0; l < 3; ++l) {
      int K = enc[s][l].Cin * 9;
      int Kp = ((K + 31) / 32) * 32;
      woff[s][l] = wcur;
      wcur += (size_t)enc[s][l].Cout * Kp;
    }
  _Float16* WPACK = (_Float16*)take(wcur * sizeof(_Float16));
  const size_t MAXACT = (size_t)BATCH * 128 * 128 * 128;
  _Float16* BUFA = (_Float16*)take(MAXACT * sizeof(_Float16));
  _Float16* BUFB = (_Float16*)take(MAXACT * sizeof(_Float16));
  _Float16* INSR = (_Float16*)take((size_t)BATCH * NINST * 128 * 128 * sizeof(_Float16));
  float* AREA  = (float*)take(BATCH * NINST * sizeof(float));
  float* MURS  = (float*)take(BATCH * 4 * 2 * sizeof(float));
  float* P4    = (float*)take((size_t)BATCH * NINST * 128 * sizeof(float));
  float* P3    = (float*)take((size_t)BATCH * NINST * 128 * sizeof(float));
  float* P2    = (float*)take((size_t)BATCH * NINST * 128 * sizeof(float));
  float* HB    = (float*)take((size_t)BATCH * NINST * 32 * sizeof(float));
  float* MIXB  = (float*)take((size_t)BATCH * NINST * 128 * sizeof(float));
  float* YB    = (float*)take((size_t)BATCH * NINST * 256 * sizeof(float));
  float* ZB    = (float*)take((size_t)BATCH * NINST * 256 * sizeof(float));
  float* LOGIT = (float*)take(BATCH * NINST * sizeof(float));

  const int BLK = 256;

  // ---- pack weights ------------------------------------------------------
  for (int s = 0; s < 3; ++s)
    for (int l = 0; l < 3; ++l) {
      int K = enc[s][l].Cin * 9;
      int Ksteps = (K + 31) / 32;
      long long total = (long long)(enc[s][l].Cout / 16) * Ksteps * 512;
      k_pack_w<<<gridFor(total, BLK), BLK, 0, stream>>>(enc[s][l].w, WPACK + woff[s][l],
                                                        enc[s][l].Cout, enc[s][l].Cin);
    }

  // ---- per-scale encoder + proto ----------------------------------------
  const int Hs[3] = {32, 64, 128};
  float* Pout[3] = {P4, P3, P2};
  for (int s = 0; s < 3; ++s) {
    int H = Hs[s], W = Hs[s], N = H * W;
    k_cvt_f16<<<gridFor((long long)BATCH * cin0[s] * N, BLK), BLK, 0, stream>>>(
        corr[s], BUFB, BATCH * cin0[s] * N);
    for (int l = 0; l < 3; ++l) {
      ConvP& cp = enc[s][l];
      long long tiles = (long long)BATCH * (cp.Cout / 16) * (N / 32);
      k_conv_wmma<<<gridFor(tiles, 8), BLK, 0, stream>>>(
          BUFB, WPACK + woff[s][l], cp.b, BUFA, BATCH, cp.Cin, cp.Cout, H, W);
      k_gn_stats<<<BATCH * 4, BLK, 0, stream>>>(BUFA, MURS, cp.Cout, N);
      k_gn_apply<<<gridFor((long long)BATCH * cp.Cout * N, BLK), BLK, 0, stream>>>(
          BUFA, MURS, cp.g, cp.be, BUFB, BATCH, cp.Cout, N);
    }
    k_resize<<<gridFor((long long)BATCH * NINST * N, BLK), BLK, 0, stream>>>(
        instances, INSR, H, W);
    k_area<<<BATCH * NINST, BLK, 0, stream>>>(INSR, AREA, N);
    k_proto_wmma<<<dim3(BATCH), BLK, 0, stream>>>(INSR, BUFB, AREA, Pout[s], N);
  }

  // ---- head: bottlenecks + MLP ------------------------------------------
  k_linear<<<gridFor(BATCH * NINST * 32, BLK), BLK, 0, stream>>>(
      P4, P3, w1_43, b1_43, HB, BATCH, NINST, 128, 32, 1);
  k_linear<<<gridFor(BATCH * NINST * 128, BLK), BLK, 0, stream>>>(
      HB, nullptr, w2_43, b2_43, MIXB, BATCH, NINST, 32, 128, 1);
  k_linear<<<gridFor(BATCH * NINST * 32, BLK), BLK, 0, stream>>>(
      MIXB, P2, w1_32, b1_32, HB, BATCH, NINST, 128, 32, 1);
  k_linear<<<gridFor(BATCH * NINST * 128, BLK), BLK, 0, stream>>>(
      HB, nullptr, w2_32, b2_32, MIXB, BATCH, NINST, 32, 128, 1);
  k_linear<<<gridFor(BATCH * NINST * 256, BLK), BLK, 0, stream>>>(
      MIXB, nullptr, wd0, bd0, YB, BATCH, NINST, 128, 256, 1);
  k_linear<<<gridFor(BATCH * NINST * 256, BLK), BLK, 0, stream>>>(
      YB, nullptr, wd1, bd1, ZB, BATCH, NINST, 256, 256, 1);

  float* out_cats  = (float*)d_out;                    // (8,1,64)
  float* out_masks = (float*)d_out + BATCH * NINST;    // (8,2,256,256)
  k_logit<<<gridFor(BATCH * NINST, BLK), BLK, 0, stream>>>(ZB, wd2, bd2, LOGIT, out_cats, 256);
  k_mask<<<gridFor((long long)BATCH * SRC_H * SRC_W, BLK), BLK, 0, stream>>>(
      LOGIT, instances, out_masks);
}
